// EAGNN_14121852469804
// MI455X (gfx1250) — matrix operand order repaired and checked by reading, not compiled, not run on hardware
//
#include <hip/hip_runtime.h>
#include <hip/hip_bf16.h>
#include <math.h>

// ---------------------------------------------------------------------------
// Types for CDNA5 WMMA (wave32): v_wmma_f32_16x16x32_bf16
// ---------------------------------------------------------------------------
typedef __bf16  v16bf __attribute__((ext_vector_type(16)));
typedef float   v8f   __attribute__((ext_vector_type(8)));

#define HDIM 128

// A-fragment (16x32 bf16, doc layout): lane<16 -> row=lane, K = {0..7, 16..23};
// lane>=16 -> row=lane-16, K = {8..15, 24..31}. Two 16B loads per lane.
__device__ __forceinline__ v16bf load_a_frag(const __bf16* rowbase, int k0, int lane) {
    const int kb = (lane >> 4) * 8;
    const __bf16* p = rowbase + k0 + kb;
    v16bf a;
    ((uint4*)&a)[0] = *(const uint4*)(p);
    ((uint4*)&a)[1] = *(const uint4*)(p + 16);
    return a;
}

// B-fragment (32x16 bf16): lane<16 -> col=lane, K = {0..7, 16..23}; lane>=16 -> col=lane-16,
// K = {8..15, 24..31} (packed 2/VGPR).  Weights pre-transposed col-major: Wt[n*K + k].
__device__ __forceinline__ v16bf load_b_frag(const __bf16* Wt, int K, int n, int k0, int lane) {
    const __bf16* p = Wt + (size_t)n * K + k0 + ((lane >> 4) << 3);
    v16bf b;
    ((uint4*)&b)[0] = *(const uint4*)(p);
    ((uint4*)&b)[1] = *(const uint4*)(p + 16);
    return b;
}

#define WMMA_BF16(a, b, c) \
    __builtin_amdgcn_wmma_f32_16x16x32_bf16(false, (a), false, (b), (short)0, (c), false, false)

__device__ __forceinline__ float silu(float v) { return v / (1.f + __expf(-v)); }

// ---------------------------------------------------------------------------
// Weight transpose+convert:  Wt[n*K + k] = (bf16) W[k*Nc + n]
// ---------------------------------------------------------------------------
__global__ void __launch_bounds__(256) convw_kernel(const float* W, __bf16* Wt, int K, int Nc) {
    int idx = blockIdx.x * 256 + threadIdx.x;
    if (idx < K * Nc) {
        int n = idx / K, k = idx % K;
        Wt[idx] = (__bf16)W[k * Nc + n];
    }
}

// ---------------------------------------------------------------------------
// In-degree
// ---------------------------------------------------------------------------
__global__ void __launch_bounds__(256) deg_kernel(const long long* ei, float* deg, long long E) {
    long long e = (long long)blockIdx.x * 256 + threadIdx.x;
    if (e < E) unsafeAtomicAdd(&deg[(int)ei[E + e]], 1.0f);
}

// ---------------------------------------------------------------------------
// Node encoder: h = silu(x@W1+b1)@W2 + b2   (3->128 scalar, 128->128 WMMA)
// 16 nodes/block.
// ---------------------------------------------------------------------------
__global__ void __launch_bounds__(256) node_enc_kernel(
    const float* x, const float* W1, const float* b1,
    const __bf16* Wt2, const float* b2, float* h, __bf16* hB)
{
    __shared__ __align__(16) __bf16 Hbuf[16 * HDIM];
    const int t = threadIdx.x, trow = t >> 4, tcol = t & 15;
    const long long node = (long long)blockIdx.x * 16 + trow;
    const float x0 = x[node * 3], x1 = x[node * 3 + 1], x2 = x[node * 3 + 2];
    #pragma unroll
    for (int q = 0; q < 8; ++q) {
        int c = tcol * 8 + q;
        float v = x0 * W1[c] + x1 * W1[HDIM + c] + x2 * W1[2 * HDIM + c] + b1[c];
        Hbuf[trow * HDIM + c] = (__bf16)silu(v);
    }
    __syncthreads();
    const int lane = t & 31, wv = t >> 5;
    const int row = lane & 15, n = wv * 16 + row, mb = (lane >> 4) * 8;
    const __bf16* arow = &Hbuf[row * HDIM];
    v8f acc = {};
    #pragma unroll
    for (int ks = 0; ks < 4; ++ks) {
        v16bf a = load_a_frag(arow, ks * 32, lane);
        v16bf b = load_b_frag(Wt2, HDIM, n, ks * 32, lane);
        acc = WMMA_BF16(a, b, acc);
    }
    const float bias = b2[n];
    #pragma unroll
    for (int r = 0; r < 8; ++r) {
        long long m = (long long)blockIdx.x * 16 + mb + r;
        float v = acc[r] + bias;
        h[m * HDIM + n]  = v;
        hB[m * HDIM + n] = (__bf16)v;
    }
}

// ---------------------------------------------------------------------------
// Edge encoder: edge_attr = silu(ef@W1+b1)@W2 + b2, ef = [rel, |rel|]
// 32 edges/block; each wave computes two 16x16 tiles sharing B fragments.
// ---------------------------------------------------------------------------
__global__ void __launch_bounds__(256) edge_enc_kernel(
    const float* coords, const long long* ei, long long E,
    const float* W1, const float* b1,
    const __bf16* Wt2, const float* b2, __bf16* eaB)
{
    __shared__ __align__(16) __bf16 Hbuf[32 * HDIM];
    const int t = threadIdx.x, trow = t >> 3, tc8 = t & 7;  // 32 rows, 8 thr/row
    const long long e = (long long)blockIdx.x * 32 + trow;
    const long long s = ei[e], d = ei[E + e];
    const float dx = coords[d * 3]     - coords[s * 3];
    const float dy = coords[d * 3 + 1] - coords[s * 3 + 1];
    const float dz = coords[d * 3 + 2] - coords[s * 3 + 2];
    const float dn = sqrtf(dx * dx + dy * dy + dz * dz);
    #pragma unroll
    for (int q = 0; q < 16; ++q) {
        int c = tc8 * 16 + q;
        float v = dx * W1[c] + dy * W1[HDIM + c] + dz * W1[2 * HDIM + c]
                + dn * W1[3 * HDIM + c] + b1[c];
        Hbuf[trow * HDIM + c] = (__bf16)silu(v);
    }
    __syncthreads();
    const int lane = t & 31, wv = t >> 5;
    const int row = lane & 15, n = wv * 16 + row, mb = (lane >> 4) * 8;
    const __bf16* arow0 = &Hbuf[row * HDIM];
    const __bf16* arow1 = &Hbuf[(row + 16) * HDIM];
    v8f acc0 = {}, acc1 = {};
    #pragma unroll
    for (int ks = 0; ks < 4; ++ks) {
        v16bf b  = load_b_frag(Wt2, HDIM, n, ks * 32, lane);
        v16bf a0 = load_a_frag(arow0, ks * 32, lane);
        v16bf a1 = load_a_frag(arow1, ks * 32, lane);
        acc0 = WMMA_BF16(a0, b, acc0);
        acc1 = WMMA_BF16(a1, b, acc1);
    }
    const float bias = b2[n];
    #pragma unroll
    for (int r = 0; r < 8; ++r) {
        long long m = (long long)blockIdx.x * 32 + mb + r;
        eaB[m * HDIM + n]               = (__bf16)(acc0[r] + bias);
        eaB[(m + 16) * HDIM + n]        = (__bf16)(acc1[r] + bias);
    }
}

// ---------------------------------------------------------------------------
// Edge message: msg = silu([h_dst|h_src|ea]@We1+b)@We2+b, scatter-add to agg.
// 32 edges/block, 8 waves each own a 16-col strip and compute two M-tiles,
// reusing each B fragment for 2 WMMAs (halves L2 weight traffic).
// K=384 then K=128.
// ---------------------------------------------------------------------------
__global__ void __launch_bounds__(256) edge_msg_kernel(
    const long long* ei, long long E,
    const __bf16* hB, const __bf16* eaB,
    const __bf16* Wt1, const float* b1,
    const __bf16* Wt2, const float* b2, float* agg)
{
    __shared__ __align__(16) __bf16 Abuf[32 * 384];   // 24 KB
    __shared__ __align__(16) __bf16 Hbuf[32 * HDIM];  //  8 KB
    __shared__ int dsts[32];
    const int t = threadIdx.x, trow = t >> 3, tc8 = t & 7;  // 32 rows, 8 thr/row
    const long long e0 = (long long)blockIdx.x * 32;
    const long long s = ei[e0 + trow], d = ei[E + e0 + trow];
    if (tc8 == 0) dsts[trow] = (int)d;
    #pragma unroll
    for (int j = 0; j < 6; ++j) {                     // 48 chunks of 8 bf16 / row
        int ci = tc8 + j * 8;
        const __bf16* sp;
        if (ci < 16)      sp = hB  + d * HDIM + ci * 8;
        else if (ci < 32) sp = hB  + s * HDIM + (ci - 16) * 8;
        else              sp = eaB + (e0 + trow) * HDIM + (ci - 32) * 8;
        *(uint4*)&Abuf[trow * 384 + ci * 8] = *(const uint4*)sp;
    }
    __syncthreads();
    const int lane = t & 31, wv = t >> 5;
    const int row = lane & 15, n = wv * 16 + row, mb = (lane >> 4) * 8;
    const __bf16* arow0 = &Abuf[row * 384];
    const __bf16* arow1 = &Abuf[(row + 16) * 384];
    v8f acc0 = {}, acc1 = {};
    #pragma unroll
    for (int ks = 0; ks < 12; ++ks) {                 // K = 384
        v16bf b  = load_b_frag(Wt1, 384, n, ks * 32, lane);
        v16bf a0 = load_a_frag(arow0, ks * 32, lane);
        v16bf a1 = load_a_frag(arow1, ks * 32, lane);
        acc0 = WMMA_BF16(a0, b, acc0);
        acc1 = WMMA_BF16(a1, b, acc1);
    }
    const float bias = b1[n];
    #pragma unroll
    for (int r = 0; r < 8; ++r) {
        Hbuf[(mb + r) * HDIM + n]      = (__bf16)silu(acc0[r] + bias);
        Hbuf[(16 + mb + r) * HDIM + n] = (__bf16)silu(acc1[r] + bias);
    }
    __syncthreads();
    const __bf16* hrow0 = &Hbuf[row * HDIM];
    const __bf16* hrow1 = &Hbuf[(row + 16) * HDIM];
    v8f out0 = {}, out1 = {};
    #pragma unroll
    for (int ks = 0; ks < 4; ++ks) {                  // K = 128
        v16bf b  = load_b_frag(Wt2, HDIM, n, ks * 32, lane);
        v16bf a0 = load_a_frag(hrow0, ks * 32, lane);
        v16bf a1 = load_a_frag(hrow1, ks * 32, lane);
        out0 = WMMA_BF16(a0, b, out0);
        out1 = WMMA_BF16(a1, b, out1);
    }
    const float bias2 = b2[n];
    #pragma unroll
    for (int r = 0; r < 8; ++r) {
        unsafeAtomicAdd(&agg[(size_t)dsts[mb + r]      * HDIM + n], out0[r] + bias2);
        unsafeAtomicAdd(&agg[(size_t)dsts[16 + mb + r] * HDIM + n], out1[r] + bias2);
    }
}

// ---------------------------------------------------------------------------
// Node update: upd = silu([h | agg/denom]@Wn1+b)@Wn2+b   (K=256, K=128)
// 16 nodes/block.
// ---------------------------------------------------------------------------
__global__ void __launch_bounds__(256) node_upd_kernel(
    const __bf16* hB, const float* agg, const float* deg,
    const __bf16* Wt1, const float* b1,
    const __bf16* Wt2, const float* b2, float* upd)
{
    __shared__ __align__(16) __bf16 Abuf[16 * 256];
    __shared__ __align__(16) __bf16 Hbuf[16 * HDIM];
    const int t = threadIdx.x, trow = t >> 4, tcol = t & 15;
    const long long node = (long long)blockIdx.x * 16 + trow;
    const float rden = 1.f / fmaxf(deg[node], 1.f);
    #pragma unroll
    for (int j = 0; j < 2; ++j) {                     // 32 chunks of 8 per row
        int ci = tcol + j * 16;
        if (ci < 16) {
            *(uint4*)&Abuf[trow * 256 + ci * 8] = *(const uint4*)(hB + node * HDIM + ci * 8);
        } else {
            const float* ap = agg + node * HDIM + (ci - 16) * 8;
            #pragma unroll
            for (int q = 0; q < 8; ++q)
                Abuf[trow * 256 + ci * 8 + q] = (__bf16)(ap[q] * rden);
        }
    }
    __syncthreads();
    const int lane = t & 31, wv = t >> 5;
    const int row = lane & 15, n = wv * 16 + row, mb = (lane >> 4) * 8;
    const __bf16* arow = &Abuf[row * 256];
    v8f acc = {};
    #pragma unroll
    for (int ks = 0; ks < 8; ++ks) {                  // K = 256
        v16bf a = load_a_frag(arow, ks * 32, lane);
        v16bf b = load_b_frag(Wt1, 256, n, ks * 32, lane);
        acc = WMMA_BF16(a, b, acc);
    }
    const float bias = b1[n];
    #pragma unroll
    for (int r = 0; r < 8; ++r)
        Hbuf[(mb + r) * HDIM + n] = (__bf16)silu(acc[r] + bias);
    __syncthreads();
    const __bf16* hrow = &Hbuf[row * HDIM];
    v8f acc2 = {};
    #pragma unroll
    for (int ks = 0; ks < 4; ++ks) {
        v16bf a = load_a_frag(hrow, ks * 32, lane);
        v16bf b = load_b_frag(Wt2, HDIM, n, ks * 32, lane);
        acc2 = WMMA_BF16(a, b, acc2);
    }
    const float bias2 = b2[n];
    #pragma unroll
    for (int r = 0; r < 8; ++r) {
        long long m = (long long)blockIdx.x * 16 + mb + r;
        upd[m * HDIM + n] = acc2[r] + bias2;
    }
}

// ---------------------------------------------------------------------------
// Residual + LayerNorm: h = LN(upd + h) * g + b ; also refresh bf16 copy
// ---------------------------------------------------------------------------
__global__ void __launch_bounds__(128) ln_kernel(
    const float* upd, float* h, __bf16* hB, const float* g, const float* bb)
{
    __shared__ float red[HDIM];
    __shared__ float s_mean, s_rstd;
    const long long nidx = blockIdx.x;
    const int c = threadIdx.x;
    const float s = upd[nidx * HDIM + c] + h[nidx * HDIM + c];
    red[c] = s; __syncthreads();
    for (int o = 64; o > 0; o >>= 1) { if (c < o) red[c] += red[c + o]; __syncthreads(); }
    if (c == 0) s_mean = red[0] * (1.f / HDIM);
    __syncthreads();
    const float dv = s - s_mean;
    red[c] = dv * dv; __syncthreads();
    for (int o = 64; o > 0; o >>= 1) { if (c < o) red[c] += red[c + o]; __syncthreads(); }
    if (c == 0) s_rstd = rsqrtf(red[0] * (1.f / HDIM) + 1e-5f);
    __syncthreads();
    const float o = dv * s_rstd * g[c] + bb[c];
    h[nidx * HDIM + c]  = o;
    hB[nidx * HDIM + c] = (__bf16)o;
}

// ---------------------------------------------------------------------------
// Decoder: out = silu(h@Wd1+b)@Wd2 + b   (WMMA 128x128, then scalar 128->9)
// ---------------------------------------------------------------------------
__global__ void __launch_bounds__(256) dec_kernel(
    const __bf16* hB, const __bf16* Wt1, const float* b1,
    const float* W2, const float* b2, float* out)
{
    __shared__ __align__(16) float Hbuf[16 * HDIM];
    const int t = threadIdx.x;
    const int lane = t & 31, wv = t >> 5;
    const int row = lane & 15, n = wv * 16 + row, mb = (lane >> 4) * 8;
    const long long node = (long long)blockIdx.x * 16 + row;
    const __bf16* arow = hB + node * HDIM;            // A direct from global (L2)
    v8f acc = {};
    #pragma unroll
    for (int ks = 0; ks < 4; ++ks) {
        v16bf a = load_a_frag(arow, ks * 32, lane);
        v16bf b = load_b_frag(Wt1, HDIM, n, ks * 32, lane);
        acc = WMMA_BF16(a, b, acc);
    }
    const float bias = b1[n];
    #pragma unroll
    for (int r = 0; r < 8; ++r)
        Hbuf[(mb + r) * HDIM + n] = silu(acc[r] + bias);
    __syncthreads();
    if (t < 144) {                                    // 16 rows x 9 cols
        int r = t / 9, c = t % 9;
        float s2 = b2[c];
        for (int k = 0; k < HDIM; ++k) s2 += Hbuf[r * HDIM + k] * W2[k * 9 + c];
        out[((long long)blockIdx.x * 16 + r) * 9 + c] = s2;
    }
}

// ---------------------------------------------------------------------------
// Host orchestration
// ---------------------------------------------------------------------------
extern "C" void kernel_launch(void* const* d_in, const int* in_sizes, int n_in,
                              void* d_out, int out_size, void* d_ws, size_t ws_size,
                              hipStream_t stream) {
    const float*      x      = (const float*)d_in[0];
    const float*      coords = (const float*)d_in[1];
    const long long*  ei     = (const long long*)d_in[2];
    const float *W_ne1=(const float*)d_in[3],  *b_ne1=(const float*)d_in[4];
    const float *W_ne2=(const float*)d_in[5],  *b_ne2=(const float*)d_in[6];
    const float *W_ee1=(const float*)d_in[7],  *b_ee1=(const float*)d_in[8];
    const float *W_ee2=(const float*)d_in[9],  *b_ee2=(const float*)d_in[10];
    const float *We1=(const float*)d_in[11],   *be1=(const float*)d_in[12];
    const float *We2=(const float*)d_in[13],   *be2=(const float*)d_in[14];
    const float *Wn1=(const float*)d_in[15],   *bn1=(const float*)d_in[16];
    const float *Wn2=(const float*)d_in[17],   *bn2=(const float*)d_in[18];
    const float *ln_g=(const float*)d_in[19],  *ln_b=(const float*)d_in[20];
    const float *Wd1=(const float*)d_in[21],   *bd1=(const float*)d_in[22];
    const float *Wd2=(const float*)d_in[23],   *bd2=(const float*)d_in[24];

    const long long N = in_sizes[0] / 3;
    const long long E = in_sizes[2] / 2;
    const int       L = in_sizes[12] / HDIM;

    char* ws = (char*)d_ws;
    size_t off = 0;
    auto take = [&](size_t bytes) { size_t o = off; off += (bytes + 255) & ~(size_t)255; return o; };
    float*  h    = (float*) (ws + take((size_t)N * HDIM * 4));
    __bf16* hB   = (__bf16*)(ws + take((size_t)N * HDIM * 2));
    float*  agg  = (float*) (ws + take((size_t)N * HDIM * 4));
    float*  upd  = (float*) (ws + take((size_t)N * HDIM * 4));
    float*  deg  = (float*) (ws + take((size_t)N * 4));
    __bf16* eaB  = (__bf16*)(ws + take((size_t)E * HDIM * 2));
    __bf16* WtNe2= (__bf16*)(ws + take(128 * 128 * 2));
    __bf16* WtEe2= (__bf16*)(ws + take(128 * 128 * 2));
    __bf16* WtE1 = (__bf16*)(ws + take((size_t)L * 128 * 384 * 2));
    __bf16* WtE2 = (__bf16*)(ws + take((size_t)L * 128 * 128 * 2));
    __bf16* WtN1 = (__bf16*)(ws + take((size_t)L * 128 * 256 * 2));
    __bf16* WtN2 = (__bf16*)(ws + take((size_t)L * 128 * 128 * 2));
    __bf16* WtD1 = (__bf16*)(ws + take(128 * 128 * 2));

    auto conv = [&](const float* W, __bf16* Wt, int K, int Nc) {
        int tot = K * Nc;
        convw_kernel<<<(tot + 255) / 256, 256, 0, stream>>>(W, Wt, K, Nc);
    };
    conv(W_ne2, WtNe2, 128, 128);
    conv(W_ee2, WtEe2, 128, 128);
    conv(Wd1,   WtD1,  128, 128);
    for (int l = 0; l < L; ++l) {
        conv(We1 + (size_t)l * 384 * 128, WtE1 + (size_t)l * 128 * 384, 384, 128);
        conv(We2 + (size_t)l * 128 * 128, WtE2 + (size_t)l * 128 * 128, 128, 128);
        conv(Wn1 + (size_t)l * 256 * 128, WtN1 + (size_t)l * 128 * 256, 256, 128);
        conv(Wn2 + (size_t)l * 128 * 128, WtN2 + (size_t)l * 128 * 128, 128, 128);
    }

    hipMemsetAsync(deg, 0, (size_t)N * 4, stream);
    deg_kernel<<<(int)((E + 255) / 256), 256, 0, stream>>>(ei, deg, E);

    node_enc_kernel<<<(int)(N / 16), 256, 0, stream>>>(x, W_ne1, b_ne1, WtNe2, b_ne2, h, hB);
    edge_enc_kernel<<<(int)(E / 32), 256, 0, stream>>>(coords, ei, E, W_ee1, b_ee1, WtEe2, b_ee2, eaB);

    for (int l = 0; l < L; ++l) {
        hipMemsetAsync(agg, 0, (size_t)N * HDIM * 4, stream);
        edge_msg_kernel<<<(int)(E / 32), 256, 0, stream>>>(
            ei, E, hB, eaB,
            WtE1 + (size_t)l * 128 * 384, be1 + (size_t)l * HDIM,
            WtE2 + (size_t)l * 128 * 128, be2 + (size_t)l * HDIM, agg);
        node_upd_kernel<<<(int)(N / 16), 256, 0, stream>>>(
            hB, agg, deg,
            WtN1 + (size_t)l * 128 * 256, bn1 + (size_t)l * HDIM,
            WtN2 + (size_t)l * 128 * 128, bn2 + (size_t)l * HDIM, upd);
        ln_kernel<<<(int)N, 128, 0, stream>>>(upd, h, hB,
            ln_g + (size_t)l * HDIM, ln_b + (size_t)l * HDIM);
    }

    dec_kernel<<<(int)(N / 16), 256, 0, stream>>>(hB, WtD1, bd1, Wd2, bd2, (float*)d_out);
}
